// PEGCN_72095321031133
// MI455X (gfx1250) — matrix operand order repaired and checked by reading, not compiled, not run on hardware
//
#include <hip/hip_runtime.h>
#include <hip/hip_bf16.h>
#include <math.h>

typedef __attribute__((ext_vector_type(16))) _Float16 v16h;
typedef __attribute__((ext_vector_type(8)))  _Float16 v8h;
typedef __attribute__((ext_vector_type(8)))  float    v8f;

#define S_H  136           // f16 LDS row stride: 272B -> bank offset 4m mod 64, conflict-free; 16B-aligned runs
#define WPB  4             // waves per block for GEMM kernels

static __device__ __forceinline__ v8f wmma_f16(v16h a, v16h b, v8f c) {
    return __builtin_amdgcn_wmma_f32_16x16x32_f16(false, a, false, b, (short)0, c, false, false);
}

// A-fragment (16x32 f16) from f16 LDS activations, per ISA 05_wmma layout:
// lane m=L%16, kh=L/16; halves [0..7] = K k0+8kh+0..7, halves [8..15] = K k0+8kh+16..23
static __device__ __forceinline__ v16h load_a_lds(const _Float16* actrow, int m, int kh, int k0) {
    const _Float16* p = actrow + m * S_H + k0 + 8 * kh;
    v8h lo = *reinterpret_cast<const v8h*>(p);        // ds_load_b128
    v8h hi = *reinterpret_cast<const v8h*>(p + 16);   // ds_load_b128
    return __builtin_shufflevector(lo, hi, 0, 1, 2, 3, 4, 5, 6, 7, 8, 9, 10, 11, 12, 13, 14, 15);
}

// B-fragment (32x16 f16): lane L holds column N=n0+L%16; halves [0..7] = K k0+8kh+0..7, [8..15] = K k0+8kh+16..23
static __device__ __forceinline__ v16h load_b_g(const _Float16* W, int Nw, int k0, int n0, int lane) {
    v16h b;
    const _Float16* p = W + (size_t)(k0 + 8 * (lane >> 4)) * Nw + n0 + (lane & 15);
#pragma unroll
    for (int s = 0; s < 8; ++s) { b[s] = p[(size_t)s * Nw]; b[8 + s] = p[(size_t)(16 + s) * Nw]; }
    return b;
}

// One dense layer for a 16-node tile: K = KT*32 cols of LDS starting at kcol0, N = NT*16.
// A-frags hoisted to registers before emits -> in-place LDS output is safe within the wave.
template<int NT, int KT, typename F>
static __device__ __forceinline__ void dense_layer(const _Float16* actrow, int kcol0,
                                                   const _Float16* W, int Nw,
                                                   const float* bias, int lane, F emit) {
    const int m = lane & 15, kh = lane >> 4;
    v16h af[KT];
#pragma unroll
    for (int ks = 0; ks < KT; ++ks) af[ks] = load_a_lds(actrow, m, kh, kcol0 + ks * 32);
#pragma unroll
    for (int nt = 0; nt < NT; ++nt) {
        v8f c = {};
#pragma unroll
        for (int ks = 0; ks < KT; ++ks) {
            v16h b = load_b_g(W, Nw, ks * 32, nt * 16, lane);
            c = wmma_f16(af[ks], b, c);
        }
        const int n = nt * 16 + m;                 // C/D: N = lane&15
        const float bv = bias ? bias[n] : 0.0f;
#pragma unroll
        for (int r = 0; r < 8; ++r) emit(r + 8 * kh, n, c[r] + bv);   // M = r + 8*(lane/16)
    }
}

static __device__ __forceinline__ void layernorm128(_Float16* actrow, const float* g, const float* b, int lane) {
    const int m = lane & 15, kh = lane >> 4;
    _Float16* p = actrow + m * S_H + kh * 64;
    float s = 0.f, s2 = 0.f;
    for (int f = 0; f < 64; ++f) { float v = (float)p[f]; s += v; s2 += v * v; }
    s  += __shfl_xor(s, 16);
    s2 += __shfl_xor(s2, 16);
    const float mean = s * (1.0f / 128.0f);
    const float var  = s2 * (1.0f / 128.0f) - mean * mean;
    const float rs   = rsqrtf(var + 1e-5f);
    for (int f = 0; f < 64; ++f) {
        int col = kh * 64 + f;
        p[f] = (_Float16)(((float)p[f] - mean) * rs * g[col] + b[col]);
    }
}

// ---------------- prep: fp32 -> f16 weights (cw1 K-padded 19->32) ----------------
__global__ void k_wconv(const float* fw1, const float* fw2, const float* fw3,
                        const float* dw0, const float* dw1, const float* dw2,
                        const float* cw1, const float* cw2, _Float16* dst) {
    int t = blockIdx.x * blockDim.x + threadIdx.x;
    if (t >= 66048) return;
    float v;
    if      (t < 16384) v = fw1[t];
    else if (t < 32768) v = fw2[t - 16384];
    else if (t < 49152) v = fw3[t - 32768];
    else if (t < 57344) v = dw0[t - 49152];
    else if (t < 59392) v = dw1[t - 57344];
    else if (t < 59904) v = dw2[t - 59392];
    else if (t < 61952) { int i = t - 59904; int k = i / 64, n = i % 64; v = (k < 19) ? cw1[k * 64 + n] : 0.0f; }
    else                v = cw2[t - 61952];
    dst[t] = (_Float16)v;
}

// ---------------- degree / dinv ----------------
__global__ void k_fill1(float* p, int n)  { int i = blockIdx.x * blockDim.x + threadIdx.x; if (i < n) p[i] = 1.0f; }
__global__ void k_degacc(const int* row, float* deg, int E) {
    int e = blockIdx.x * blockDim.x + threadIdx.x;
    if (e < E) atomicAdd(&deg[row[e]], 1.0f);
}
__global__ void k_degfin(float* deg, int n) { int i = blockIdx.x * blockDim.x + threadIdx.x; if (i < n) deg[i] = rsqrtf(deg[i]); }
__global__ void k_zero(float* p, long long n) {
    long long i = (long long)blockIdx.x * blockDim.x + threadIdx.x;
    if (i < n) p[i] = 0.0f;
}

// ---------------- node MLP: encoder + LN + decoder + x@cw1 (all WMMA) ----------------
__global__ void __launch_bounds__(WPB * 32, 1)
k_mlp(const float* coords, const float* features,
      const float* fw0, const float* fb0, const float* fg0, const float* fe0,
      const float* fb1, const float* fg1, const float* fe1,
      const float* fb2, const float* fg2, const float* fe2,
      const float* fb3,
      const float* db0, const float* db1, const float* db2,
      const _Float16* fw1h, const _Float16* fw2h, const _Float16* fw3h,
      const _Float16* dw0h, const _Float16* dw1h, const _Float16* dw2h,
      const _Float16* cw1h,
      float* xw1, int N) {
    __shared__ _Float16 act_s[WPB][16][S_H];
    const int lane = threadIdx.x & 31;
    const int wid  = threadIdx.x >> 5;
    _Float16* actrow = &act_s[wid][0][0];
    const int g0   = (blockIdx.x * WPB + wid) * 16;
    const int m = lane & 15, kh = lane >> 4;
    const int node = g0 + m;

    // layer0: 2->128 (K too small for WMMA; VALU)
    float cx = 0.f, cy = 0.f;
    if (node < N) { cx = coords[2 * node]; cy = coords[2 * node + 1]; }
    for (int f = 0; f < 64; ++f) {
        int col = kh * 64 + f;
        float v = cx * fw0[col] + cy * fw0[128 + col] + fb0[col];
        actrow[m * S_H + col] = (_Float16)(v > 0.f ? v : 0.f);
    }
    layernorm128(actrow, fg0, fe0, lane);

    dense_layer<8, 4>(actrow, 0, fw1h, 128, fb1, lane,
        [&](int mr, int n, float v) { actrow[mr * S_H + n] = (_Float16)(v > 0.f ? v : 0.f); });
    layernorm128(actrow, fg1, fe1, lane);

    dense_layer<8, 4>(actrow, 0, fw2h, 128, fb2, lane,
        [&](int mr, int n, float v) { actrow[mr * S_H + n] = (_Float16)(v > 0.f ? v : 0.f); });
    layernorm128(actrow, fg2, fe2, lane);

    dense_layer<8, 4>(actrow, 0, fw3h, 128, fb3, lane,
        [&](int mr, int n, float v) { actrow[mr * S_H + n] = (_Float16)(v > 0.f ? v : 0.f); });

    // decoder: 128->64 (tanh, cols 0..63), 64->32 (tanh, cols 64..95), 32->16 -> cols 3..18
    dense_layer<4, 4>(actrow, 0,  dw0h, 64, db0, lane,
        [&](int mr, int n, float v) { actrow[mr * S_H + n] = (_Float16)tanhf(v); });
    dense_layer<2, 2>(actrow, 0,  dw1h, 32, db1, lane,
        [&](int mr, int n, float v) { actrow[mr * S_H + 64 + n] = (_Float16)tanhf(v); });
    dense_layer<1, 1>(actrow, 64, dw2h, 16, db2, lane,
        [&](int mr, int n, float v) { actrow[mr * S_H + 3 + n] = (_Float16)v; });

    // assemble x = [features(3) | emb(16) | zero-pad to 32] in cols 0..31
    if (kh == 0) {
        float f0 = 0.f, f1 = 0.f, f2 = 0.f;
        if (node < N) { f0 = features[3 * node]; f1 = features[3 * node + 1]; f2 = features[3 * node + 2]; }
        actrow[m * S_H + 0] = (_Float16)f0;
        actrow[m * S_H + 1] = (_Float16)f1;
        actrow[m * S_H + 2] = (_Float16)f2;
        for (int f = 19; f < 32; ++f) actrow[m * S_H + f] = (_Float16)0.f;
    }

    // xw1 = x @ cw1 (bias applied in finalize)
    dense_layer<4, 1>(actrow, 0, cw1h, 64, nullptr, lane,
        [&](int mr, int n, float v) { int nn = g0 + mr; if (nn < N) xw1[(size_t)nn * 64 + n] = v; });
}

// ---------------- GCN message passing: float4 gather + scale + atomic scatter ----------------
__global__ void k_edge(const int* row, const int* col, const float* dinv,
                       const float* src, float* dst, int E) {
    long long t = (long long)blockIdx.x * blockDim.x + threadIdx.x;
    if (t >= (long long)E * 16) return;
    int e = (int)(t >> 4), q = (int)(t & 15);
    int r = row[e], c = col[e];
    float nrm = dinv[r] * dinv[c];
    const float4 v = *reinterpret_cast<const float4*>(src + (size_t)c * 64 + q * 4);
    float* d = dst + (size_t)r * 64 + q * 4;
    atomicAdd(d + 0, v.x * nrm);
    atomicAdd(d + 1, v.y * nrm);
    atomicAdd(d + 2, v.z * nrm);
    atomicAdd(d + 3, v.w * nrm);
}

// h1 = relu(agg + xw*dinv^2 + b), written in place over agg
__global__ void k_final1(const float* xw, const float* dinv, const float* bias, float* agg, int N) {
    long long t = (long long)blockIdx.x * blockDim.x + threadIdx.x;
    if (t >= (long long)N * 64) return;
    int n = (int)(t >> 6), f = (int)(t & 63);
    float d = dinv[n];
    float v = agg[t] + xw[t] * d * d + bias[f];
    agg[t] = v > 0.f ? v : 0.f;
}

// xw2 = h1 @ cw2 (WMMA, A direct from global fp32)
__global__ void k_gemm2(const float* h1, const _Float16* cw2h, float* xw2, int N) {
    const int lane = threadIdx.x & 31, wid = threadIdx.x >> 5;
    const int g0 = (blockIdx.x * WPB + wid) * 16;
    const int m = lane & 15, kh = lane >> 4;
    const int node = g0 + m;
    v16h af[2];
#pragma unroll
    for (int ks = 0; ks < 2; ++ks) {
        v16h a;
        int k0 = ks * 32 + 8 * kh;
#pragma unroll
        for (int i = 0; i < 4; ++i) {
            int k = k0 + 2 * i;
            float v0 = 0.f, v1 = 0.f, v2 = 0.f, v3 = 0.f;
            if (node < N) {
                const float* p = h1 + (size_t)node * 64;
                v0 = p[k]; v1 = p[k + 1]; v2 = p[k + 16]; v3 = p[k + 17];
            }
            a[2*i] = (_Float16)v0;  a[2*i+1] = (_Float16)v1;
            a[8+2*i] = (_Float16)v2; a[8+2*i+1] = (_Float16)v3;
        }
        af[ks] = a;
    }
#pragma unroll
    for (int nt = 0; nt < 4; ++nt) {
        v8f c = {};
#pragma unroll
        for (int ks = 0; ks < 2; ++ks) {
            v16h b = load_b_g(cw2h, 64, ks * 32, nt * 16, lane);
            c = wmma_f16(af[ks], b, c);
        }
        int n = nt * 16 + m;
#pragma unroll
        for (int r = 0; r < 8; ++r) {
            int nn = g0 + r + 8 * kh;
            if (nn < N) xw2[(size_t)nn * 64 + n] = c[r];
        }
    }
}

// out[n] = (agg2 + xw2*dinv^2 + cb2) . ow + ob   (one wave per node)
__global__ void k_final2(const float* agg, const float* xw, const float* dinv,
                         const float* cb2, const float* ow, const float* ob, float* out, int N) {
    const int lane = threadIdx.x & 31, wid = threadIdx.x >> 5;
    const int n = blockIdx.x * 8 + wid;
    if (n >= N) return;
    const float d = dinv[n], d2 = d * d;
    float s = 0.f;
    for (int f = lane; f < 64; f += 32) {
        float v = agg[(size_t)n * 64 + f] + xw[(size_t)n * 64 + f] * d2 + cb2[f];
        s += v * ow[f];
    }
    s += __shfl_xor(s, 16); s += __shfl_xor(s, 8); s += __shfl_xor(s, 4);
    s += __shfl_xor(s, 2);  s += __shfl_xor(s, 1);
    if (lane == 0) out[n] = s + ob[0];
}

extern "C" void kernel_launch(void* const* d_in, const int* in_sizes, int n_in,
                              void* d_out, int out_size, void* d_ws, size_t ws_size,
                              hipStream_t stream) {
    const int N = in_sizes[0] / 2;
    const int E = in_sizes[2] / 2;

    const float* coords   = (const float*)d_in[0];
    const float* features = (const float*)d_in[1];
    const int*   row      = (const int*)d_in[2];
    const int*   col      = row + E;
    const float* fw0 = (const float*)d_in[3],  *fb0 = (const float*)d_in[4];
    const float* fg0 = (const float*)d_in[5],  *fe0 = (const float*)d_in[6];
    const float* fw1 = (const float*)d_in[7],  *fb1 = (const float*)d_in[8];
    const float* fg1 = (const float*)d_in[9],  *fe1 = (const float*)d_in[10];
    const float* fw2 = (const float*)d_in[11], *fb2 = (const float*)d_in[12];
    const float* fg2 = (const float*)d_in[13], *fe2 = (const float*)d_in[14];
    const float* fw3 = (const float*)d_in[15], *fb3 = (const float*)d_in[16];
    const float* dw0 = (const float*)d_in[17], *db0 = (const float*)d_in[18];
    const float* dw1 = (const float*)d_in[19], *db1 = (const float*)d_in[20];
    const float* dw2 = (const float*)d_in[21], *db2 = (const float*)d_in[22];
    const float* cw1 = (const float*)d_in[23], *cb1 = (const float*)d_in[24];
    const float* cw2 = (const float*)d_in[25], *cb2 = (const float*)d_in[26];
    const float* ow  = (const float*)d_in[27], *ob  = (const float*)d_in[28];

    // workspace layout
    char* ws = (char*)d_ws;
    size_t off = 0;
    auto take = [&](size_t bytes) -> char* {
        off = (off + 255) & ~(size_t)255;
        char* p = ws + off;
        off += bytes;
        return p;
    };
    float*    dinv = (float*)take((size_t)N * 4);
    _Float16* wh   = (_Float16*)take(66048 * 2);
    float*    B0   = (float*)take((size_t)N * 64 * 4);   // xw1, then xw2
    float*    B1   = (float*)take((size_t)N * 64 * 4);   // agg1 -> h1 -> agg2
    (void)ws_size;

    const _Float16* fw1h = wh;
    const _Float16* fw2h = wh + 16384;
    const _Float16* fw3h = wh + 32768;
    const _Float16* dw0h = wh + 49152;
    const _Float16* dw1h = wh + 57344;
    const _Float16* dw2h = wh + 59392;
    const _Float16* cw1h = wh + 59904;
    const _Float16* cw2h = wh + 61952;

    const int nb_n   = (N + 255) / 256;
    const int nb_e   = (E + 255) / 256;
    const long long NF = (long long)N * 64;
    const int nb_nf  = (int)((NF + 255) / 256);
    const int nb_e4  = (int)(((long long)E * 16 + 255) / 256);
    const int nb_t16 = (N + WPB * 16 - 1) / (WPB * 16);

    k_wconv<<<(66048 + 255) / 256, 256, 0, stream>>>(fw1, fw2, fw3, dw0, dw1, dw2, cw1, cw2, wh);

    k_fill1 <<<nb_n, 256, 0, stream>>>(dinv, N);
    k_degacc<<<nb_e, 256, 0, stream>>>(row, dinv, E);
    k_degfin<<<nb_n, 256, 0, stream>>>(dinv, N);

    k_mlp<<<nb_t16, WPB * 32, 0, stream>>>(coords, features,
        fw0, fb0, fg0, fe0, fb1, fg1, fe1, fb2, fg2, fe2, fb3,
        db0, db1, db2, fw1h, fw2h, fw3h, dw0h, dw1h, dw2h, cw1h, B0, N);

    k_zero  <<<nb_nf, 256, 0, stream>>>(B1, NF);
    k_edge  <<<nb_e4, 256, 0, stream>>>(row, col, dinv, B0, B1, E);   // agg1
    k_final1<<<nb_nf, 256, 0, stream>>>(B0, dinv, cb1, B1, N);        // B1 = h1

    k_gemm2 <<<nb_t16, WPB * 32, 0, stream>>>(B1, cw2h, B0, N);       // B0 = xw2

    k_zero  <<<nb_nf, 256, 0, stream>>>(B1, NF);
    k_edge  <<<nb_e4, 256, 0, stream>>>(row, col, dinv, B0, B1, E);   // agg2

    k_final2<<<(N + 7) / 8, 256, 0, stream>>>(B1, B0, dinv, cb2, ow, ob, (float*)d_out, N);
}
